// GraphBasedSampler_23897198035036
// MI455X (gfx1250) — matrix operand order repaired and checked by reading, not compiled, not run on hardware
//
#include <hip/hip_runtime.h>
#include <hip/hip_bf16.h>

#define EPS 1e-5f
#define CUTOFF 10.0f

typedef __attribute__((ext_vector_type(16))) __bf16 v16bf;
typedef __attribute__((ext_vector_type(8)))  __bf16 v8bf;
typedef __attribute__((ext_vector_type(8)))  float  v8f;

__device__ __forceinline__ float wave_reduce_add(float v) {
#pragma unroll
  for (int off = 16; off > 0; off >>= 1)
    v += __shfl_xor(v, off, 32);
  return v;
}

__device__ __forceinline__ v16bf cat8(v8bf lo, v8bf hi) {
  return __builtin_shufflevector(lo, hi, 0, 1, 2, 3, 4, 5, 6, 7,
                                 8, 9, 10, 11, 12, 13, 14, 15);
}

// ---------------------------------------------------------------------------
// Weight prep: transpose + quantize to bf16 so WMMA B-fragments are contiguous.
// w1: [4][384][512] -> w1T: [4][512][384] ; w2: [4][512][128] -> w2T: [4][128][512]
// ---------------------------------------------------------------------------
__global__ void k_prep_weights(const float* __restrict__ w1, const float* __restrict__ w2,
                               __bf16* __restrict__ w1T, __bf16* __restrict__ w2T) {
  const int total1 = 4 * 512 * 384;
  for (int idx = blockIdx.x * blockDim.x + threadIdx.x; idx < total1;
       idx += gridDim.x * blockDim.x) {
    int l = idx / (512 * 384), r = idx % (512 * 384);
    int n = r / 384, k = r % 384;
    w1T[idx] = (__bf16)w1[(l * 384 + k) * 512 + n];
  }
  const int total2 = 4 * 128 * 512;
  for (int idx = blockIdx.x * blockDim.x + threadIdx.x; idx < total2;
       idx += gridDim.x * blockDim.x) {
    int l = idx / (128 * 512), r = idx % (128 * 512);
    int n = r / 512, k = r % 512;
    w2T[idx] = (__bf16)w2[(l * 512 + k) * 128 + n];
  }
}

// ---------------------------------------------------------------------------
// Edge init: pairwise distance -> LN(dist*w+b) -> relu ; adjacency
// ---------------------------------------------------------------------------
__global__ void __launch_bounds__(256)
k_init_edges(const float* __restrict__ coords, const float* __restrict__ ee_w,
             const float* __restrict__ ee_b, const float* __restrict__ ee_g,
             const float* __restrict__ ee_bt, float* __restrict__ edges,
             float* __restrict__ adj) {
  const int lane = threadIdx.x & 31, w = threadIdx.x >> 5;
  const int e = blockIdx.x * 8 + w;
  const int i = e >> 8, j = e & 255;
  float dx = coords[i * 3 + 0] - coords[j * 3 + 0];
  float dy = coords[i * 3 + 1] - coords[j * 3 + 1];
  float dz = coords[i * 3 + 2] - coords[j * 3 + 2];
  float sq = dx * dx + dy * dy + dz * dz;
  float dist = sq > 0.f ? sqrtf(sq) : 0.f;
  if (lane == 0) adj[e] = (dist < CUTOFF) ? 1.f : 0.f;
  float xv[4], s = 0.f, s2 = 0.f;
#pragma unroll
  for (int t = 0; t < 4; ++t) {
    int k = lane + 32 * t;
    float x = dist * ee_w[k] + ee_b[k];
    xv[t] = x; s += x; s2 += x * x;
  }
  s = wave_reduce_add(s); s2 = wave_reduce_add(s2);
  float mean = s * (1.f / 128.f);
  float var = s2 * (1.f / 128.f) - mean * mean;
  float rstd = rsqrtf(var + EPS);
#pragma unroll
  for (int t = 0; t < 4; ++t) {
    int k = lane + 32 * t;
    float y = (xv[t] - mean) * rstd * ee_g[k] + ee_bt[k];
    edges[(size_t)e * 128 + k] = fmaxf(y, 0.f);
  }
}

// ---------------------------------------------------------------------------
// msgs[i,j] = adj[i,j] * dot(edges[i,j,:], nodes[i,:])
// ---------------------------------------------------------------------------
__global__ void __launch_bounds__(256)
k_msgs(const float* __restrict__ edges, const float* __restrict__ nodes,
       const float* __restrict__ adj, float* __restrict__ msgs) {
  __shared__ float sn[128];
  const int i = blockIdx.x;
  const int lane = threadIdx.x & 31, w = threadIdx.x >> 5;
  if (threadIdx.x < 128) sn[threadIdx.x] = nodes[i * 128 + threadIdx.x];
  __syncthreads();
  for (int j = w; j < 256; j += 8) {
    const float* er = edges + ((size_t)i * 256 + j) * 128;
    float s = 0.f;
#pragma unroll
    for (int t = 0; t < 4; ++t) { int k = lane + 32 * t; s += er[k] * sn[k]; }
    s = wave_reduce_add(s);
    if (lane == 0) msgs[i * 256 + j] = adj[i * 256 + j] * s;
  }
}

// ---------------------------------------------------------------------------
// Node update MLP (small, M=256): nd_new = nd + relu(LN([nd|msgs]@w1+b1))@w2+b2
// ---------------------------------------------------------------------------
__global__ void __launch_bounds__(256)
k_node_update(const float* __restrict__ nodes_in, const float* __restrict__ msgs,
              const float* __restrict__ w1, const float* __restrict__ b1,
              const float* __restrict__ g, const float* __restrict__ bt,
              const float* __restrict__ w2, const float* __restrict__ b2,
              float* __restrict__ nodes_out) {
  __shared__ float sNI[384];
  __shared__ float sH[512];
  __shared__ float sRed[256], sRed2[256];
  const int m = blockIdx.x, t = threadIdx.x;
  for (int c = t; c < 384; c += 256)
    sNI[c] = (c < 128) ? nodes_in[m * 128 + c] : msgs[m * 256 + (c - 128)];
  __syncthreads();
  float h0 = b1[t], h1 = b1[t + 256];
  for (int k = 0; k < 384; ++k) {
    float x = sNI[k];
    h0 += x * w1[k * 512 + t];
    h1 += x * w1[k * 512 + t + 256];
  }
  sRed[t] = h0 + h1; sRed2[t] = h0 * h0 + h1 * h1;
  __syncthreads();
  for (int off = 128; off > 0; off >>= 1) {
    if (t < off) { sRed[t] += sRed[t + off]; sRed2[t] += sRed2[t + off]; }
    __syncthreads();
  }
  float mean = sRed[0] * (1.f / 512.f);
  float var = sRed2[0] * (1.f / 512.f) - mean * mean;
  float rstd = rsqrtf(var + EPS);
  sH[t]       = fmaxf((h0 - mean) * rstd * g[t] + bt[t], 0.f);
  sH[t + 256] = fmaxf((h1 - mean) * rstd * g[t + 256] + bt[t + 256], 0.f);
  __syncthreads();
  if (t < 128) {
    float s = b2[t];
    for (int h = 0; h < 512; ++h) s += sH[h] * w2[h * 128 + t];
    nodes_out[m * 128 + t] = nodes_in[m * 128 + t] + s;
  }
}

// ---------------------------------------------------------------------------
// Fused edge-update MLP: bf16 WMMA, hidden kept in LDS (bf16, LN in place).
// Block = 256 threads (8 waves), handles 64 edge rows (all share node i).
// 8 waves = 4 row-tiles x 2 column-groups. A-fragments hoisted to registers.
// ---------------------------------------------------------------------------
__global__ void __launch_bounds__(256)
k_edge_update(float* __restrict__ edges, const float* __restrict__ nodes,
              const __bf16* __restrict__ w1T, const float* __restrict__ b1,
              const float* __restrict__ g, const float* __restrict__ bt,
              const __bf16* __restrict__ w2T, const float* __restrict__ b2) {
  __shared__ __align__(16) __bf16 sA[64][392];    // A tile (16B-aligned rows)
  __shared__ __align__(16) __bf16 sHb[64][512];   // hidden bf16; LN applied in place

  const int tid = threadIdx.x;
  const int lane = tid & 31;
  const int wv = tid >> 5;
  const int e0 = blockIdx.x << 6;   // 64 rows per block
  const int i = e0 >> 8;

  // Stage A = [edges_row | nodes_i | nodes_j] -> bf16 in LDS
  for (int idx = tid; idx < 64 * 384; idx += 256) {
    int r = idx / 384, c = idx % 384;
    int e = e0 + r;
    float x;
    if (c < 128)      x = edges[(size_t)e * 128 + c];
    else if (c < 256) x = nodes[i * 128 + (c - 128)];
    else              x = nodes[(e & 255) * 128 + (c - 256)];
    sA[r][c] = (__bf16)x;
  }
  __syncthreads();

  const int m = lane & 15;
  const int hiHalf = (lane >> 4) & 1;
  const int rt = wv >> 1;        // row tile (0..3): rows rt*16 .. rt*16+15
  const int cg = wv & 1;         // column group: cols cg*256 .. cg*256+255

  // Hoist GEMM1 A-fragments (12 k-steps) into registers, reuse over 16 N-tiles
  v16bf afr[12];
  {
    const __bf16* arow = &sA[rt * 16 + m][0];
#pragma unroll
    for (int kk = 0; kk < 12; ++kk) {
      int ab = kk * 32 + (hiHalf ? 8 : 0);
      v8bf lo = *(const v8bf*)(arow + ab);
      v8bf hi = *(const v8bf*)(arow + ab + 16);
      afr[kk] = cat8(lo, hi);
    }
  }

  // GEMM1: [64,384] x [384,512] -> sHb (bf16, pre-LN)
#pragma unroll 4
  for (int nt = 0; nt < 16; ++nt) {
    const int n0 = cg * 256 + nt * 16;
    const int nn = n0 + m;
    const float bias = b1[nn];
    v8f acc = {bias, bias, bias, bias, bias, bias, bias, bias};
    const __bf16* bbase = w1T + (size_t)nn * 384;
#pragma unroll
    for (int kk = 0; kk < 12; ++kk) {
      const __bf16* bp = bbase + kk * 32 + (hiHalf ? 8 : 0);
      v16bf b = cat8(*(const v8bf*)bp, *(const v8bf*)(bp + 16));
      acc = __builtin_amdgcn_wmma_f32_16x16x32_bf16(false, afr[kk], false, b,
                                                    (short)0, acc, false, false);
    }
#pragma unroll
    for (int vv = 0; vv < 8; ++vv)
      sHb[rt * 16 + hiHalf * 8 + vv][n0 + m] = (__bf16)acc[vv];
  }
  __syncthreads();

  // LayerNorm(512) + ReLU in place (bf16); each wave handles 8 rows.
  // gamma/beta preloaded per-lane (columns are fixed per lane).
  {
    float gv[16], bv[16];
#pragma unroll
    for (int t = 0; t < 16; ++t) { gv[t] = g[lane + 32 * t]; bv[t] = bt[lane + 32 * t]; }
#pragma unroll
    for (int rr = 0; rr < 8; ++rr) {
      const int r = wv * 8 + rr;
      float xs[16], s = 0.f, s2 = 0.f;
#pragma unroll
      for (int t = 0; t < 16; ++t) {
        float x = (float)sHb[r][lane + 32 * t];
        xs[t] = x; s += x; s2 += x * x;
      }
      s = wave_reduce_add(s); s2 = wave_reduce_add(s2);
      float mean = s * (1.f / 512.f);
      float var = s2 * (1.f / 512.f) - mean * mean;
      float rstd = rsqrtf(var + EPS);
#pragma unroll
      for (int t = 0; t < 16; ++t) {
        float y = (xs[t] - mean) * rstd * gv[t] + bv[t];
        sHb[r][lane + 32 * t] = (__bf16)fmaxf(y, 0.f);
      }
    }
  }
  __syncthreads();

  // GEMM2: [64,512] x [512,128] + bias + residual
  // wave: row-tile rt, cols (cg*4 .. cg*4+3)*16. Hoist 16 A-fragments.
  {
    v16bf hfr[16];
    const __bf16* arow = &sHb[rt * 16 + m][0];
#pragma unroll
    for (int kk = 0; kk < 16; ++kk) {
      int ab = kk * 32 + (hiHalf ? 8 : 0);
      v8bf lo = *(const v8bf*)(arow + ab);
      v8bf hi = *(const v8bf*)(arow + ab + 16);
      hfr[kk] = cat8(lo, hi);
    }
#pragma unroll
    for (int t2 = 0; t2 < 4; ++t2) {
      const int n0 = (cg * 4 + t2) * 16;
      const int nn = n0 + m;
      const float bias = b2[nn];
      v8f acc = {bias, bias, bias, bias, bias, bias, bias, bias};
      const __bf16* bbase = w2T + (size_t)nn * 512;
#pragma unroll
      for (int kk = 0; kk < 16; ++kk) {
        const __bf16* bp = bbase + kk * 32 + (hiHalf ? 8 : 0);
        v16bf b = cat8(*(const v8bf*)bp, *(const v8bf*)(bp + 16));
        acc = __builtin_amdgcn_wmma_f32_16x16x32_bf16(false, hfr[kk], false, b,
                                                      (short)0, acc, false, false);
      }
#pragma unroll
      for (int vv = 0; vv < 8; ++vv) {
        int row = rt * 16 + hiHalf * 8 + vv;
        size_t off = (size_t)(e0 + row) * 128 + n0 + m;
        edges[off] += acc[vv];
      }
    }
  }
}

// ---------------------------------------------------------------------------
// Final heads: nodes_out = relu(LN(nd@no1+b))@no2+b ; coords = relu(LN(.@sp1+b))@sp2+b
// ---------------------------------------------------------------------------
__global__ void __launch_bounds__(256)
k_head(const float* __restrict__ nd,
       const float* __restrict__ no1_w, const float* __restrict__ no1_b,
       const float* __restrict__ no_g, const float* __restrict__ no_bt,
       const float* __restrict__ no2_w, const float* __restrict__ no2_b,
       const float* __restrict__ sp1_w, const float* __restrict__ sp1_b,
       const float* __restrict__ sp_g, const float* __restrict__ sp_bt,
       const float* __restrict__ sp2_w, const float* __restrict__ sp2_b,
       float* __restrict__ out) {
  __shared__ float sX[128];
  __shared__ float sH[512];
  __shared__ float sNO[128];
  __shared__ float sRed[256], sRed2[256];
  const int m = blockIdx.x, t = threadIdx.x;
  if (t < 128) sX[t] = nd[m * 128 + t];
  __syncthreads();
  float h0 = no1_b[t], h1 = no1_b[t + 256];
  for (int k = 0; k < 128; ++k) {
    float x = sX[k];
    h0 += x * no1_w[k * 512 + t];
    h1 += x * no1_w[k * 512 + t + 256];
  }
  sRed[t] = h0 + h1; sRed2[t] = h0 * h0 + h1 * h1;
  __syncthreads();
  for (int off = 128; off > 0; off >>= 1) {
    if (t < off) { sRed[t] += sRed[t + off]; sRed2[t] += sRed2[t + off]; }
    __syncthreads();
  }
  float mean = sRed[0] * (1.f / 512.f);
  float var = sRed2[0] * (1.f / 512.f) - mean * mean;
  float rstd = rsqrtf(var + EPS);
  sH[t]       = fmaxf((h0 - mean) * rstd * no_g[t] + no_bt[t], 0.f);
  sH[t + 256] = fmaxf((h1 - mean) * rstd * no_g[t + 256] + no_bt[t + 256], 0.f);
  __syncthreads();
  if (t < 128) {
    float s = no2_b[t];
    for (int h = 0; h < 512; ++h) s += sH[h] * no2_w[h * 128 + t];
    out[m * 128 + t] = s;
    sNO[t] = s;
  }
  __syncthreads();
  float g0 = sp1_b[t], g1 = sp1_b[t + 256];
  for (int k = 0; k < 128; ++k) {
    float x = sNO[k];
    g0 += x * sp1_w[k * 512 + t];
    g1 += x * sp1_w[k * 512 + t + 256];
  }
  __syncthreads();
  sRed[t] = g0 + g1; sRed2[t] = g0 * g0 + g1 * g1;
  __syncthreads();
  for (int off = 128; off > 0; off >>= 1) {
    if (t < off) { sRed[t] += sRed[t + off]; sRed2[t] += sRed2[t + off]; }
    __syncthreads();
  }
  mean = sRed[0] * (1.f / 512.f);
  var = sRed2[0] * (1.f / 512.f) - mean * mean;
  rstd = rsqrtf(var + EPS);
  __syncthreads();
  sH[t]       = fmaxf((g0 - mean) * rstd * sp_g[t] + sp_bt[t], 0.f);
  sH[t + 256] = fmaxf((g1 - mean) * rstd * sp_g[t + 256] + sp_bt[t + 256], 0.f);
  __syncthreads();
  if (t < 3) {
    float s = sp2_b[t];
    for (int h = 0; h < 512; ++h) s += sH[h] * sp2_w[h * 3 + t];
    out[32768 + m * 3 + t] = s;
  }
}

// ---------------------------------------------------------------------------
extern "C" void kernel_launch(void* const* d_in, const int* in_sizes, int n_in,
                              void* d_out, int out_size, void* d_ws, size_t ws_size,
                              hipStream_t stream) {
  const float* nodes  = (const float*)d_in[0];
  const float* coords = (const float*)d_in[1];
  const float* ee_w   = (const float*)d_in[2];
  const float* ee_b   = (const float*)d_in[3];
  const float* ee_g   = (const float*)d_in[4];
  const float* ee_bt  = (const float*)d_in[5];
  const float* nu1_w  = (const float*)d_in[6];
  const float* nu1_b  = (const float*)d_in[7];
  const float* nu_g   = (const float*)d_in[8];
  const float* nu_bt  = (const float*)d_in[9];
  const float* nu2_w  = (const float*)d_in[10];
  const float* nu2_b  = (const float*)d_in[11];
  const float* eu1_w  = (const float*)d_in[12];
  const float* eu1_b  = (const float*)d_in[13];
  const float* eu_g   = (const float*)d_in[14];
  const float* eu_bt  = (const float*)d_in[15];
  const float* eu2_w  = (const float*)d_in[16];
  const float* eu2_b  = (const float*)d_in[17];
  const float* sp1_w  = (const float*)d_in[18];
  const float* sp1_b  = (const float*)d_in[19];
  const float* sp_g   = (const float*)d_in[20];
  const float* sp_bt  = (const float*)d_in[21];
  const float* sp2_w  = (const float*)d_in[22];
  const float* sp2_b  = (const float*)d_in[23];
  const float* no1_w  = (const float*)d_in[24];
  const float* no1_b  = (const float*)d_in[25];
  const float* no_g   = (const float*)d_in[26];
  const float* no_bt  = (const float*)d_in[27];
  const float* no2_w  = (const float*)d_in[28];
  const float* no2_b  = (const float*)d_in[29];

  char* ws = (char*)d_ws;
  float*  edges  = (float*)(ws);                    // 65536*128 f32 = 32 MB
  float*  adj    = (float*)(ws + 33554432);         // 65536 f32
  float*  msgs   = (float*)(ws + 33816576);         // 256*256 f32
  float*  nodesA = (float*)(ws + 34078720);         // 256*128 f32
  float*  nodesB = (float*)(ws + 34209792);         // 256*128 f32
  __bf16* w1T    = (__bf16*)(ws + 34340864);        // 4*512*384 bf16
  __bf16* w2T    = (__bf16*)(ws + 35913728);        // 4*128*512 bf16

  k_prep_weights<<<512, 256, 0, stream>>>(eu1_w, eu2_w, w1T, w2T);
  k_init_edges<<<8192, 256, 0, stream>>>(coords, ee_w, ee_b, ee_g, ee_bt, edges, adj);

  const float* nsrc = nodes;
  for (int l = 0; l < 4; ++l) {
    float* ndst = (l & 1) ? nodesB : nodesA;
    k_msgs<<<256, 256, 0, stream>>>(edges, nsrc, adj, msgs);
    k_node_update<<<256, 256, 0, stream>>>(
        nsrc, msgs, nu1_w + l * 384 * 512, nu1_b + l * 512, nu_g + l * 512,
        nu_bt + l * 512, nu2_w + l * 512 * 128, nu2_b + l * 128, ndst);
    k_edge_update<<<1024, 256, 0, stream>>>(
        edges, nsrc, w1T + l * 512 * 384, eu1_b + l * 512, eu_g + l * 512,
        eu_bt + l * 512, w2T + l * 128 * 512, eu2_b + l * 128);
    nsrc = ndst;
  }
  k_head<<<256, 256, 0, stream>>>(nsrc, no1_w, no1_b, no_g, no_bt, no2_w, no2_b,
                                  sp1_w, sp1_b, sp_g, sp_bt, sp2_w, sp2_b,
                                  (float*)d_out);
}
